// MHSA_33603824124549
// MI455X (gfx1250) — matrix operand order repaired and checked by reading, not compiled, not run on hardware
//
#include <hip/hip_runtime.h>

typedef __attribute__((ext_vector_type(16))) _Float16 v16h;
typedef __attribute__((ext_vector_type(8)))  float    v8f;

__device__ __forceinline__ v8f wmma16(v16h a, v16h b, v8f c) {
    // D = A(16x32 f16) * B(32x16 f16) + C(16x16 f32)
    return __builtin_amdgcn_wmma_f32_16x16x32_f16(false, a, false, b, (short)0, c, false, false);
}

// A fragment: A[m][k] row-major, stride ld (elements).
// lanes 0-15: M=lane, K 0..15 ; lanes 16-31: M=lane-16, K 16..31
template <typename T>
__device__ __forceinline__ v16h load_a(const T* base, int ld) {
    int lane = threadIdx.x & 31;
    int m = lane & 15;
    int kb = (lane < 16) ? 0 : 16;
    const T* p = base + m * ld + kb;
    v16h a;
#pragma unroll
    for (int j = 0; j < 16; ++j) a[j] = (_Float16)p[j];
    return a;
}

// B fragment from pre-swizzled storage: frag[n][k] with k contiguous (32 halfs per n).
__device__ __forceinline__ v16h load_b_swz(const _Float16* base) {
    int lane = threadIdx.x & 31;
    int n = lane & 15;
    int kb = (lane < 16) ? 0 : 16;
    const _Float16* p = base + (size_t)n * 32 + kb;
    v16h b;
#pragma unroll
    for (int j = 0; j < 16; ++j) b[j] = p[j];
    return b;
}

// B fragment: B[k][n] row-major (stride ld); 16 lane-coalesced strided loads.
__device__ __forceinline__ v16h load_b_rm(const float* base, int ld) {
    int lane = threadIdx.x & 31;
    int n = lane & 15;
    int kb = (lane < 16) ? 0 : 16;
    const float* p = base + (size_t)kb * ld + n;
    v16h b;
#pragma unroll
    for (int j = 0; j < 16; ++j) b[j] = (_Float16)p[(size_t)j * ld];
    return b;
}

// B fragment from transposed storage: element (k,n) at base[n*ld + k] (k contiguous).
__device__ __forceinline__ v16h load_bt(const float* base, int ld) {
    int lane = threadIdx.x & 31;
    int n = lane & 15;
    int kb = (lane < 16) ? 0 : 16;
    const float* p = base + (size_t)n * ld + kb;
    v16h b;
#pragma unroll
    for (int j = 0; j < 16; ++j) b[j] = (_Float16)p[j];
    return b;
}

// ------ Kernel 0: fold BN params into per-channel scale/shift (1024 floats) ---------------
__global__ __launch_bounds__(256) void k_bnprep(
        const float* __restrict__ g1, const float* __restrict__ b1,
        const float* __restrict__ m1, const float* __restrict__ v1,
        const float* __restrict__ g2, const float* __restrict__ b2,
        const float* __restrict__ m2, const float* __restrict__ v2,
        float* __restrict__ st) {  // [s_qk 256][t_qk 256][s_v 256][t_v 256]
    const int i = blockIdx.x * 256 + threadIdx.x;
    if (i < 256) {
        float s = g1[i] * rsqrtf(v1[i] + 1e-5f);
        st[i] = s;
        st[256 + i] = b1[i] - m1[i] * s;
    } else {
        int j = i - 256;
        float s = g2[j] * rsqrtf(v2[j] + 1e-5f);
        st[512 + j] = s;
        st[768 + j] = b2[j] - m2[j] * s;
    }
}

// ------ Kernel 1: qk = BN(Wqk @ x) per window; 2 M-tiles/wave share the B load ------------
__global__ __launch_bounds__(32) void k_qk_conv(
        const float* __restrict__ x, const float* __restrict__ W,
        const float* __restrict__ st,
        _Float16* __restrict__ qh, _Float16* __restrict__ kh) {
    const int pt = blockIdx.x, ct = blockIdx.y, win = blockIdx.z;  // ct: 32-channel group
    const int lane = threadIdx.x & 31;
    const int hi = lane >> 4, ln = lane & 15;
    const int p0 = pt * 16, ch0 = ct * 32;
    const int p = p0 + ln;
    v8f acc[2] = {};
    if (pt != 12) {  // uniform: all 16 columns valid, mask-free path
        for (int k0 = 0; k0 < 256; k0 += 32) {
            v16h a0 = load_a(W + (size_t)ch0 * 256 + k0, 256);
            v16h a1 = load_a(W + (size_t)(ch0 + 16) * 256 + k0, 256);
            v16h b;
            const int kb = hi ? 16 : 0;
            const float* Bp = x + (size_t)win * 256 * 196 + (size_t)(k0 + kb) * 196 + p;
#pragma unroll
            for (int j = 0; j < 16; ++j) b[j] = (_Float16)Bp[(size_t)j * 196];
            acc[0] = wmma16(a0, b, acc[0]);  // independent accumulators: no WMMA RAW pair
            acc[1] = wmma16(a1, b, acc[1]);
        }
    } else {  // edge tile: clamp address, zero masked columns
        const float pmask = (p < 196) ? 1.f : 0.f;
        const int pc = (p < 196) ? p : 195;
        for (int k0 = 0; k0 < 256; k0 += 32) {
            v16h a0 = load_a(W + (size_t)ch0 * 256 + k0, 256);
            v16h a1 = load_a(W + (size_t)(ch0 + 16) * 256 + k0, 256);
            v16h b;
            const int kb = hi ? 16 : 0;
            const float* Bp = x + (size_t)win * 256 * 196 + (size_t)(k0 + kb) * 196 + pc;
#pragma unroll
            for (int j = 0; j < 16; ++j) b[j] = (_Float16)(Bp[(size_t)j * 196] * pmask);
            acc[0] = wmma16(a0, b, acc[0]);
            acc[1] = wmma16(a1, b, acc[1]);
        }
        if (p >= 196) return;
    }
    if (ct < 4) {  // uniform per block: q half (channels < 128)
#pragma unroll
        for (int t2 = 0; t2 < 2; ++t2)
#pragma unroll
            for (int r = 0; r < 8; ++r) {
                int ch = ch0 + t2 * 16 + r + hi * 8;
                float val = acc[t2][r] * st[ch] + st[256 + ch];
                int h = ch >> 4, d = ch & 15;
                qh[((size_t)(win * 8 + h) * 208 + p) * 32 + d] = (_Float16)val;
            }
    } else {  // k half: swizzled for B-fragments [win][h][ntile=13][n=16][k=32]
#pragma unroll
        for (int t2 = 0; t2 < 2; ++t2)
#pragma unroll
            for (int r = 0; r < 8; ++r) {
                int ch = ch0 + t2 * 16 + r + hi * 8;
                float val = acc[t2][r] * st[ch] + st[256 + ch];
                int c2 = ch - 128, h = c2 >> 4, d = c2 & 15;
                kh[(((size_t)(win * 8 + h) * 13 + (p >> 4)) * 16 + (p & 15)) * 32 + d] =
                    (_Float16)val;
            }
    }
}

// ------ Kernel 2: vv = BN(Wv @ v_in); 2 M-tiles/wave share the B load ---------------------
__global__ __launch_bounds__(32) void k_v_conv(
        const float* __restrict__ vin, const float* __restrict__ W,
        const float* __restrict__ st, _Float16* __restrict__ vh) {
    const int pt = blockIdx.x, ct = blockIdx.y;
    const int lane = threadIdx.x & 31;
    const int hi = lane >> 4, ln = lane & 15;
    const int p0 = pt * 16, ch0 = ct * 32;
    v8f acc[2] = {};
    for (int k0 = 0; k0 < 256; k0 += 32) {
        v16h a0 = load_a(W + (size_t)ch0 * 256 + k0, 256);
        v16h a1 = load_a(W + (size_t)(ch0 + 16) * 256 + k0, 256);
        v16h b = load_b_rm(vin + (size_t)k0 * 3136 + p0, 3136);
        acc[0] = wmma16(a0, b, acc[0]);
        acc[1] = wmma16(a1, b, acc[1]);
    }
    const int p = p0 + ln;
    const int y = p / 56, xx = p % 56;
    const int win = (y / 14) * 4 + (xx / 14);
    const int key = (y % 14) * 14 + (xx % 14);
    const int kt = key >> 5, kk = key & 31;
#pragma unroll
    for (int t2 = 0; t2 < 2; ++t2)
#pragma unroll
        for (int r = 0; r < 8; ++r) {
            int ch = ch0 + t2 * 16 + r + hi * 8;
            float val = acc[t2][r] * st[512 + ch] + st[768 + ch];
            // vh swizzled for B-fragments: [h][win][ktile=7][dtile=2][n=16][k=32]
            int h = ch >> 5, dv = ch & 31, dt = dv >> 4, d = dv & 15;
            vh[(((((size_t)(h * 16 + win) * 7 + kt) * 2 + dt) * 16 + d) * 32) + kk] =
                (_Float16)val;
        }
}

// ------ Kernel 3: S = q k^T / 4 + bias, softmax rows, store att_win (f32) -----------------
template <bool QEDGE, bool KEDGE>
__device__ __forceinline__ void store_tile(float* sm, const v8f& c, int q0, int hi, int ln,
                                           int nt, int h, const float* __restrict__ pos_emb,
                                           const int* __restrict__ rel_idx) {
    const int k = nt * 16 + ln;
    const int kc = KEDGE ? ((k < 196) ? k : 195) : k;
#pragma unroll
    for (int r = 0; r < 8; ++r) {
        int M = r + hi * 8;
        int q = q0 + M;
        int qc = QEDGE ? ((q < 196) ? q : 195) : q;
        float bias = pos_emb[h * 729 + rel_idx[qc * 196 + kc]];  // unconditional gather
        float val = c[r] * 0.25f + bias;
        if (QEDGE || KEDGE) {
            bool ok = (!QEDGE || q < 196) && (!KEDGE || k < 196);
            val = ok ? val : -3.0e38f;
        }
        sm[M * 208 + k] = val;
    }
}

__global__ __launch_bounds__(32) void k_attn(
        const _Float16* __restrict__ qh, const _Float16* __restrict__ kh,
        const float* __restrict__ pos_emb, const int* __restrict__ rel_idx,
        float* __restrict__ att_win) {
    __shared__ float sm[16 * 208];
    const int qt = blockIdx.x, h = blockIdx.y, win = blockIdx.z;
    const int lane = threadIdx.x & 31;
    const int hi = lane >> 4, ln = lane & 15;
    const int q0 = qt * 16;
    v16h a = load_a(qh + ((size_t)(win * 8 + h) * 208 + q0) * 32, 32);
    const _Float16* Kb = kh + (size_t)(win * 8 + h) * 13 * 16 * 32;
    if (qt != 12) {  // uniform: all 16 q rows valid
        for (int nt = 0; nt < 12; ++nt) {
            v8f c = {};
            c = wmma16(a, load_b_swz(Kb + (size_t)nt * 16 * 32), c);
            store_tile<false, false>(sm, c, q0, hi, ln, nt, h, pos_emb, rel_idx);
        }
        v8f c = {};
        c = wmma16(a, load_b_swz(Kb + (size_t)12 * 16 * 32), c);
        store_tile<false, true>(sm, c, q0, hi, ln, 12, h, pos_emb, rel_idx);
    } else {
        for (int nt = 0; nt < 12; ++nt) {
            v8f c = {};
            c = wmma16(a, load_b_swz(Kb + (size_t)nt * 16 * 32), c);
            store_tile<true, false>(sm, c, q0, hi, ln, nt, h, pos_emb, rel_idx);
        }
        v8f c = {};
        c = wmma16(a, load_b_swz(Kb + (size_t)12 * 16 * 32), c);
        store_tile<true, true>(sm, c, q0, hi, ln, 12, h, pos_emb, rel_idx);
    }
    __syncthreads();
    // softmax: each row handled by a lane pair (lane, lane+16), 104 cols each
    {
        const int row = ln;
        float* rp = sm + row * 208 + hi * 104;
        float mx = -3.4e38f;
        for (int j = 0; j < 104; ++j) mx = fmaxf(mx, rp[j]);
        mx = fmaxf(mx, __shfl_xor(mx, 16));
        float ssum = 0.f;
        for (int j = 0; j < 104; ++j) {
            float e = __expf(rp[j] - mx);
            rp[j] = e;
            ssum += e;
        }
        ssum += __shfl_xor(ssum, 16);
        float inv = 1.f / ssum;
        float* dst = att_win + ((size_t)(win * 8 + h) * 208 + q0 + row) * 224 + hi * 104;
        for (int j = 0; j < 104; ++j) dst[j] = rp[j] * inv;
    }
}

// ------ Kernel 4: m = att_win @ v_win; both d-tiles/wave share the f32 A load -------------
__global__ __launch_bounds__(32) void k_av(
        const float* __restrict__ att_win, const _Float16* __restrict__ vh,
        float* __restrict__ m_out) {
    const int qt = blockIdx.x, h = blockIdx.y, win = blockIdx.z;
    const int lane = threadIdx.x & 31;
    const int hi = lane >> 4, ln = lane & 15;
    v8f acc[2] = {};
    const float* Ab = att_win + ((size_t)(win * 8 + h) * 208 + qt * 16) * 224;
    const _Float16* Bb = vh + (size_t)(h * 16 + win) * 7 * 2 * 16 * 32;
    for (int kt = 0; kt < 7; ++kt) {
        v16h a = load_a(Ab + kt * 32, 224);
        v16h b0 = load_b_swz(Bb + (size_t)kt * 2 * 16 * 32);
        v16h b1 = load_b_swz(Bb + (size_t)kt * 2 * 16 * 32 + 16 * 32);
        acc[0] = wmma16(a, b0, acc[0]);
        acc[1] = wmma16(a, b1, acc[1]);
    }
#pragma unroll
    for (int t2 = 0; t2 < 2; ++t2)
#pragma unroll
        for (int r = 0; r < 8; ++r)
            m_out[((size_t)(win * 8 + h) * 208 + qt * 16 + r + hi * 8) * 32 + t2 * 16 + ln] =
                acc[t2][r];
}

// ------ Kernel 5: expand att_full = att_win * ca_fac -> d_out (315 MB stream) -------------
__global__ __launch_bounds__(256) void k_expand(
        const float* __restrict__ att_win, const float* __restrict__ ca,
        float* __restrict__ att_out) {
    __shared__ float caf[16];
    const int Q = blockIdx.x, h = blockIdx.y;
    const int qy = Q / 56, qx = Q % 56;
    const int t = threadIdx.x;
    const int q_in = (qy % 14) * 14 + (qx % 14);
    if (t < 16) {
        int knh = t >> 2, knw = t & 3;
        int i = (qy >> 1) * 28 + (qx >> 1);
        caf[t] = ca[((size_t)h * 784 + i) * 784 + knh * 196 + ((qy >> 1) % 7) * 28 + knw * 7 +
                    ((qx >> 1) % 7)];
        // pull this block's 16 att_win rows (L2-resident) toward the WGP
        __builtin_prefetch(att_win + ((size_t)(t * 8 + h) * 208 + q_in) * 224, 0, 3);
    }
    __syncthreads();
    float* dst = att_out + ((size_t)h * 3136 + Q) * 3136;
    for (int k = t; k < 3136; k += 256) {
        int ky = k / 56, kx = k % 56;
        int wr = ky / 14, wc = kx / 14;
        float aval = att_win[((size_t)((wr * 4 + wc) * 8 + h) * 208 + q_in) * 224 +
                             (ky % 14) * 14 + (kx % 14)];
        dst[k] = aval * caf[wr * 4 + wc];
    }
}

// ------ Kernel 6: out_pre[p][h*32+d] = sum_win ca_fac * m ---------------------------------
__global__ __launch_bounds__(256) void k_combine(
        const float* __restrict__ m_in, const float* __restrict__ ca,
        float* __restrict__ out_pre) {
    const int idx = blockIdx.x * 256 + threadIdx.x;  // < 802816
    const int ch = idx & 255, p = idx >> 8;
    const int h = ch >> 5, d = ch & 31;
    const int qy = p / 56, qx = p % 56;
    const int q_in = (qy % 14) * 14 + (qx % 14);
    const int i = (qy >> 1) * 28 + (qx >> 1);
    const size_t cab = ((size_t)h * 784 + i) * 784 + ((qy >> 1) % 7) * 28 + ((qx >> 1) % 7);
    float acc = 0.f;
#pragma unroll
    for (int w = 0; w < 16; ++w) {
        float cf = ca[cab + (w >> 2) * 196 + (w & 3) * 7];
        acc += cf * m_in[((size_t)(w * 8 + h) * 208 + q_in) * 32 + d];
    }
    out_pre[idx] = acc;
}

// ------ Kernel 7: out = out_pre @ fc_w^T + fc_b; 2 N-tiles/wave share the A load ----------
__global__ __launch_bounds__(32) void k_fc(
        const float* __restrict__ out_pre, const float* __restrict__ fc_w,
        const float* __restrict__ fc_b, float* __restrict__ out) {
    const int pt = blockIdx.x, ot = blockIdx.y;  // ot: 32-output group
    const int lane = threadIdx.x & 31;
    const int hi = lane >> 4, ln = lane & 15;
    const int o0 = ot * 32;
    v8f acc[2] = {};
    for (int k0 = 0; k0 < 256; k0 += 32) {
        v16h a = load_a(out_pre + (size_t)(pt * 16) * 256 + k0, 256);
        v16h b0 = load_bt(fc_w + (size_t)o0 * 256 + k0, 256);
        v16h b1 = load_bt(fc_w + (size_t)(o0 + 16) * 256 + k0, 256);
        acc[0] = wmma16(a, b0, acc[0]);
        acc[1] = wmma16(a, b1, acc[1]);
    }
#pragma unroll
    for (int t2 = 0; t2 < 2; ++t2) {
        int o = o0 + t2 * 16 + ln;
        float bias = fc_b[o];
#pragma unroll
        for (int r = 0; r < 8; ++r) {
            int p = pt * 16 + r + hi * 8;
            out[(size_t)p * 256 + o] = acc[t2][r] + bias;
        }
    }
}

extern "C" void kernel_launch(void* const* d_in, const int* in_sizes, int n_in,
                              void* d_out, int out_size, void* d_ws, size_t ws_size,
                              hipStream_t stream) {
    const float* x        = (const float*)d_in[0];
    const float* v_in     = (const float*)d_in[1];
    const float* ca       = (const float*)d_in[2];
    const float* wqk_w    = (const float*)d_in[3];
    const float* qk_gamma = (const float*)d_in[4];
    const float* qk_beta  = (const float*)d_in[5];
    const float* qk_mean  = (const float*)d_in[6];
    const float* qk_var   = (const float*)d_in[7];
    const float* wv_w     = (const float*)d_in[8];
    const float* v_gamma  = (const float*)d_in[9];
    const float* v_beta   = (const float*)d_in[10];
    const float* v_mean   = (const float*)d_in[11];
    const float* v_var    = (const float*)d_in[12];
    const float* fc_w     = (const float*)d_in[13];
    const float* fc_b     = (const float*)d_in[14];
    const float* pos_emb  = (const float*)d_in[15];
    const int*   rel_idx  = (const int*)d_in[16];
    // d_in[17] = ca_index: index math is hardcoded; not needed.

    float* out     = (float*)d_out;
    float* att_out = out + 802816;  // out (1,56,56,256) then att (1,8,3136,3136)

    char* ws = (char*)d_ws;
    _Float16* qh      = (_Float16*)(ws + 0);          // [16][8][208][32] f16
    _Float16* kh      = (_Float16*)(ws + 1703936);    // [16][8][13][16][32] f16 (B-swizzled)
    _Float16* vh      = (_Float16*)(ws + 3407872);    // [8][16][7][2][16][32] f16 (B-swizzled)
    float*    att_win = (float*)(ws + 5242880);       // [16][8][208][224] f32
    float*    m_buf   = (float*)(ws + 29097984);      // [16][8][208][32] f32
    float*    out_pre = (float*)(ws + 32505856);      // [3136][256] f32
    float*    bn_st   = (float*)(ws + 35717120);      // [4][256] folded BN scale/shift

    // Zero-fill once per launch so all WMMA K/N padding regions are valid zeros.
    hipMemsetAsync(d_ws, 0, 35717120, stream);

    k_bnprep<<<2, 256, 0, stream>>>(qk_gamma, qk_beta, qk_mean, qk_var,
                                    v_gamma, v_beta, v_mean, v_var, bn_st);
    k_qk_conv<<<dim3(13, 8, 16), 32, 0, stream>>>(x, wqk_w, bn_st, qh, kh);
    k_v_conv<<<dim3(196, 8), 32, 0, stream>>>(v_in, wv_w, bn_st, vh);
    k_attn<<<dim3(13, 8, 16), 32, 0, stream>>>(qh, kh, pos_emb, rel_idx, att_win);
    k_av<<<dim3(13, 8, 16), 32, 0, stream>>>(att_win, vh, m_buf);
    k_expand<<<dim3(3136, 8), 256, 0, stream>>>(att_win, ca, att_out);
    k_combine<<<3136, 256, 0, stream>>>(m_buf, ca, out_pre);
    k_fc<<<dim3(196, 8), 32, 0, stream>>>(out_pre, fc_w, fc_b, out);
}
